// MultiHeadCrossAttention_25305947308842
// MI455X (gfx1250) — matrix operand order repaired and checked by reading, not compiled, not run on hardware
//
#include <hip/hip_runtime.h>

// Problem constants (match reference)
#define B_   4
#define T_   2048
#define S_   2048
#define DQ_  512
#define DKV_ 1024
#define H_   8
#define HD_  64

typedef __attribute__((ext_vector_type(16))) __bf16        v16bf;
typedef __attribute__((ext_vector_type(8)))  float         v8f;
typedef __attribute__((ext_vector_type(4)))  unsigned int  u32x4;
typedef __attribute__((ext_vector_type(4)))  float         f32x4;
typedef unsigned short u16;
typedef long long i64;

// fp32 -> bf16 round-to-nearest-even
__device__ __forceinline__ u16 f2bf(float f) {
  unsigned int u = __float_as_uint(f);
  u += 0x7fffu + ((u >> 16) & 1u);
  return (u16)(u >> 16);
}
__device__ __forceinline__ unsigned int pack2(float lo, float hi) {
  return ((unsigned int)f2bf(hi) << 16) | (unsigned int)f2bf(lo);
}

union FragU { v16bf bf; u32x4 q[2]; unsigned int u[8]; };

// A fragment (16x32 bf16, M x K): lane holds one row; p already points at
// &A[m*lda + k0 + (lane>>4)*8]. VGPR0..3 <- k..k+7, VGPR4..7 <- k+16..k+23.
__device__ __forceinline__ v16bf ldfragA(const u16* p) {
  FragU f;
  f.q[0] = *(const u32x4*)p;
  f.q[1] = *(const u32x4*)(p + 16);
  return f.bf;
}
// B fragment (32x16 bf16, K x N) from a transposed [N,K] matrix: lane holds one
// column; p points at &Bt[n*ldb + k0 + (lane>>4)*8]; VGPR0..3 <- k..k+7, VGPR4..7 <- k+16..k+23.
__device__ __forceinline__ v16bf ldfragB(const u16* p) {
  FragU f;
  f.q[0] = *(const u32x4*)p;
  f.q[1] = *(const u32x4*)(p + 16);
  return f.bf;
}
// A fragment from fp32 source (converted in-register)
__device__ __forceinline__ v16bf ldfragA_f32(const float* p) {
  f32x4 f0 = *(const f32x4*)p;
  f32x4 f1 = *(const f32x4*)(p + 4);
  f32x4 f2 = *(const f32x4*)(p + 16);
  f32x4 f3 = *(const f32x4*)(p + 20);
  FragU fr;
  fr.u[0] = pack2(f0[0], f0[1]); fr.u[1] = pack2(f0[2], f0[3]);
  fr.u[2] = pack2(f1[0], f1[1]); fr.u[3] = pack2(f1[2], f1[3]);
  fr.u[4] = pack2(f2[0], f2[1]); fr.u[5] = pack2(f2[2], f2[3]);
  fr.u[6] = pack2(f3[0], f3[1]); fr.u[7] = pack2(f3[2], f3[3]);
  return fr.bf;
}

#define WMMA_BF16(a, b, c) \
  __builtin_amdgcn_wmma_f32_16x16x32_bf16(false, (a), false, (b), (short)0, (c), false, false)

// ---------------------------------------------------------------- converts --
__global__ __launch_bounds__(256) void cvt_bf16(const float* __restrict__ in,
                                                unsigned int* __restrict__ out,
                                                i64 n2) {
  i64 i = (i64)blockIdx.x * 256 + threadIdx.x;
  if (i < n2) out[i] = pack2(in[2 * i], in[2 * i + 1]);
}

// W [K,N] fp32 -> WT [N,K] bf16
__global__ __launch_bounds__(256) void cvt_transpose(const float* __restrict__ W,
                                                     u16* __restrict__ WT,
                                                     int K, int N) {
  i64 id = (i64)blockIdx.x * 256 + threadIdx.x;
  if (id >= (i64)K * N) return;
  int k = (int)(id / N), n = (int)(id % N);
  WT[(i64)n * K + k] = f2bf(W[id]);
}

// --------------------------------------------------------- generic bf16 GEMM
// C[M,N] = A[M,Kdim] @ Bt[N,Kdim]^T + bias.  Block = 8 waves; wave -> 16x64
// tile; block -> 128x64.  Manually unrolled x2 ping-pong pipeline: no buffer
// rotation register moves, loads of the next k-step overlap the WMMAs of the
// current one.  Requires Kdim % 64 == 0 (true for all call sites).
// mode: 0 = fp32 out, 1 = bf16 out, 2 = bf16 out transposed per batch
// (out[b][n][s], rpb rows per batch, ldo = N).
__global__ __launch_bounds__(256) void gemm_bf16_nt(
    const u16* __restrict__ A, int lda,
    const u16* __restrict__ Bt, int ldb,
    const float* __restrict__ bias, int Kdim,
    void* __restrict__ out, int mode, int ldo, int rpb) {
  const int w = threadIdx.x >> 5, lane = threadIdx.x & 31;
  const int lidx = lane & 15, lhalf = lane >> 4;
  const int m0 = blockIdx.x * 128 + w * 16;
  const int n0 = blockIdx.y * 64;
  const int m = m0 + lidx;

  const u16* ap = A + (i64)m * lda + lhalf * 8;
  const u16* bp[4];
#pragma unroll
  for (int j = 0; j < 4; ++j)
    bp[j] = Bt + (i64)(n0 + 16 * j + lidx) * ldb + lhalf * 8;

  v8f acc[4] = {};
  v16bf a0 = ldfragA(ap);
  v16bf b0[4], b1[4];
  v16bf a1;
#pragma unroll
  for (int j = 0; j < 4; ++j) b0[j] = ldfragB(bp[j]);

  int k0 = 0;
  for (; k0 + 64 < Kdim; k0 += 64) {
    a1 = ldfragA(ap + k0 + 32);
#pragma unroll
    for (int j = 0; j < 4; ++j) b1[j] = ldfragB(bp[j] + k0 + 32);
#pragma unroll
    for (int j = 0; j < 4; ++j) acc[j] = WMMA_BF16(a0, b0[j], acc[j]);
    a0 = ldfragA(ap + k0 + 64);
#pragma unroll
    for (int j = 0; j < 4; ++j) b0[j] = ldfragB(bp[j] + k0 + 64);
#pragma unroll
    for (int j = 0; j < 4; ++j) acc[j] = WMMA_BF16(a1, b1[j], acc[j]);
  }
  // epilogue: final two k-steps (k0 and k0+32 == Kdim-32)
  a1 = ldfragA(ap + k0 + 32);
#pragma unroll
  for (int j = 0; j < 4; ++j) b1[j] = ldfragB(bp[j] + k0 + 32);
#pragma unroll
  for (int j = 0; j < 4; ++j) acc[j] = WMMA_BF16(a0, b0[j], acc[j]);
#pragma unroll
  for (int j = 0; j < 4; ++j) acc[j] = WMMA_BF16(a1, b1[j], acc[j]);

#pragma unroll
  for (int j = 0; j < 4; ++j) {
    const int n = n0 + 16 * j + lidx;
    const float bb = bias ? bias[n] : 0.0f;
#pragma unroll
    for (int v = 0; v < 8; ++v) {
      const int mr = m0 + lhalf * 8 + v;   // C/D layout: VGPR v -> M = v (+8 hi)
      float val = acc[j][v] + bb;
      if (mode == 0) {
        ((float*)out)[(i64)mr * ldo + n] = val;
      } else if (mode == 1) {
        ((u16*)out)[(i64)mr * ldo + n] = f2bf(val);
      } else {
        const int bat = mr / rpb, s = mr % rpb;
        ((u16*)out)[(i64)bat * ldo * rpb + (i64)n * rpb + s] = f2bf(val);
      }
    }
  }
}

// ------------------------------------------------------------ attn scores --
// scores[b,h,t,s] = (Q[b,t,h,:] . K[b,s,h,:]) / 8 + attn_mask[t,s]  (pad->-inf)
// Kdim = HD = 64: both k-steps' fragments are loaded in one burst, then the 8
// WMMAs issue back-to-back.
__global__ __launch_bounds__(256) void attn_scores(
    const u16* __restrict__ Q, const u16* __restrict__ Kp,
    const float* __restrict__ amask, const int* __restrict__ kpm,
    float* __restrict__ attn) {
  const int w = threadIdx.x >> 5, lane = threadIdx.x & 31;
  const int lidx = lane & 15, lhalf = lane >> 4;
  const int bh = blockIdx.z, b = bh >> 3, h = bh & 7;
  const int t0 = blockIdx.x * 128 + w * 16;
  const int s0 = blockIdx.y * 64;
  const u16* Qb = Q + (i64)b * T_ * DQ_ + h * HD_;
  const u16* Kb = Kp + (i64)b * S_ * DQ_ + h * HD_;
  const int t = t0 + lidx;

  const u16* aq = Qb + (i64)t * DQ_ + lhalf * 8;
  v16bf a0 = ldfragA(aq);
  v16bf a1 = ldfragA(aq + 32);
  v16bf b0[4], b1[4];
#pragma unroll
  for (int j = 0; j < 4; ++j) {
    const u16* kb = Kb + (i64)(s0 + 16 * j + lidx) * DQ_ + lhalf * 8;
    b0[j] = ldfragB(kb);
    b1[j] = ldfragB(kb + 32);
  }
  v8f acc[4] = {};
#pragma unroll
  for (int j = 0; j < 4; ++j) acc[j] = WMMA_BF16(a0, b0[j], acc[j]);
#pragma unroll
  for (int j = 0; j < 4; ++j) acc[j] = WMMA_BF16(a1, b1[j], acc[j]);

  const float NEG = -__builtin_inff();
  float* arow = attn + (i64)bh * T_ * S_;
#pragma unroll
  for (int j = 0; j < 4; ++j) {
    const int s = s0 + 16 * j + lidx;
    const bool masked = (kpm[(i64)b * S_ + s] == 0);
#pragma unroll
    for (int v = 0; v < 8; ++v) {
      const int tr = t0 + lhalf * 8 + v;
      float val = acc[j][v] * 0.125f + amask[(i64)tr * S_ + s];
      if (masked) val = NEG;
      arow[(i64)tr * S_ + s] = val;
    }
  }
}

// ---------------------------------------------------------------- softmax --
// One block per row; 8 elements/thread held in registers: 1 read + 1 write.
__global__ __launch_bounds__(256) void softmax_row(float* __restrict__ attn) {
  float* p = attn + (i64)blockIdx.x * S_;
  const int tid = threadIdx.x;
  float x[8];
#pragma unroll
  for (int i = 0; i < 8; ++i) x[i] = p[tid + 256 * i];
  float mx = x[0];
#pragma unroll
  for (int i = 1; i < 8; ++i) mx = fmaxf(mx, x[i]);
  __shared__ float red[256];
  red[tid] = mx; __syncthreads();
  for (int off = 128; off > 0; off >>= 1) {
    if (tid < off) red[tid] = fmaxf(red[tid], red[tid + off]);
    __syncthreads();
  }
  mx = red[0];
  __syncthreads();
  const bool ok = (mx > -3.0e38f);
  float sum = 0.f;
#pragma unroll
  for (int i = 0; i < 8; ++i) {
    x[i] = ok ? __expf(x[i] - mx) : 0.f;
    sum += x[i];
  }
  red[tid] = sum; __syncthreads();
  for (int off = 128; off > 0; off >>= 1) {
    if (tid < off) red[tid] += red[tid + off];
    __syncthreads();
  }
  const float inv = (red[0] > 0.f) ? 1.0f / red[0] : 0.f;
#pragma unroll
  for (int i = 0; i < 8; ++i) p[tid + 256 * i] = x[i] * inv;
}

// ---------------------------------------------------------------- P @ V -----
// O[b,t,h,:] = attn[b,h,t,:] @ V[b,:,h,:]  using V stored transposed [B,DQ,S].
// Same x2 ping-pong pipeline; fp32 attn rows are converted to bf16 in
// registers, overlapping the packing VALU work with the WMMA pipe.
__global__ __launch_bounds__(256) void attn_value(
    const float* __restrict__ attn, const u16* __restrict__ Vt,
    u16* __restrict__ Obf) {
  const int w = threadIdx.x >> 5, lane = threadIdx.x & 31;
  const int lidx = lane & 15, lhalf = lane >> 4;
  const int bh = blockIdx.y, b = bh >> 3, h = bh & 7;
  const int t0 = blockIdx.x * 128 + w * 16;
  const float* Pb = attn + (i64)bh * T_ * S_;
  const u16* Vb = Vt + (i64)b * DQ_ * S_ + (i64)(h * HD_) * S_;
  const int t = t0 + lidx;
  const float* aprow = Pb + (i64)t * S_ + lhalf * 8;
  const u16* bp[4];
#pragma unroll
  for (int j = 0; j < 4; ++j)
    bp[j] = Vb + (i64)(16 * j + lidx) * S_ + lhalf * 8;

  v8f acc[4] = {};
  v16bf a0 = ldfragA_f32(aprow);
  v16bf b0[4], b1[4];
  v16bf a1;
#pragma unroll
  for (int j = 0; j < 4; ++j) b0[j] = ldfragB(bp[j]);

  int k0 = 0;
  for (; k0 + 64 < S_; k0 += 64) {
    a1 = ldfragA_f32(aprow + k0 + 32);
#pragma unroll
    for (int j = 0; j < 4; ++j) b1[j] = ldfragB(bp[j] + k0 + 32);
#pragma unroll
    for (int j = 0; j < 4; ++j) acc[j] = WMMA_BF16(a0, b0[j], acc[j]);
    a0 = ldfragA_f32(aprow + k0 + 64);
#pragma unroll
    for (int j = 0; j < 4; ++j) b0[j] = ldfragB(bp[j] + k0 + 64);
#pragma unroll
    for (int j = 0; j < 4; ++j) acc[j] = WMMA_BF16(a1, b1[j], acc[j]);
  }
  a1 = ldfragA_f32(aprow + k0 + 32);
#pragma unroll
  for (int j = 0; j < 4; ++j) b1[j] = ldfragB(bp[j] + k0 + 32);
#pragma unroll
  for (int j = 0; j < 4; ++j) acc[j] = WMMA_BF16(a0, b0[j], acc[j]);
#pragma unroll
  for (int j = 0; j < 4; ++j) acc[j] = WMMA_BF16(a1, b1[j], acc[j]);

#pragma unroll
  for (int j = 0; j < 4; ++j) {
    const int n = 16 * j + lidx;
#pragma unroll
    for (int v = 0; v < 8; ++v) {
      const int tr = t0 + lhalf * 8 + v;
      Obf[((i64)b * T_ + tr) * DQ_ + h * HD_ + n] = f2bf(acc[j][v]);
    }
  }
}

// ---------------------------------------------------------------- launch ----
extern "C" void kernel_launch(void* const* d_in, const int* in_sizes, int n_in,
                              void* d_out, int out_size, void* d_ws, size_t ws_size,
                              hipStream_t stream) {
  const float* query = (const float*)d_in[0];
  const float* key   = (const float*)d_in[1];
  const float* value = (const float*)d_in[2];
  const int*   kpm   = (const int*)d_in[3];
  const float* amask = (const float*)d_in[4];
  const float* Wq = (const float*)d_in[5];  const float* bq = (const float*)d_in[6];
  const float* Wk = (const float*)d_in[7];  const float* bk = (const float*)d_in[8];
  const float* Wv = (const float*)d_in[9];  const float* bv = (const float*)d_in[10];
  const float* Wo = (const float*)d_in[11]; const float* bo = (const float*)d_in[12];

  float* out  = (float*)d_out;                       // [B,T,DQ]
  float* attn = out + (i64)B_ * T_ * DQ_;            // [B,H,T,S]

  char* p = (char*)d_ws;
  u16* q_bf = (u16*)p; p += (i64)B_ * T_ * DQ_  * 2;
  u16* kin  = (u16*)p; p += (i64)B_ * S_ * DKV_ * 2;
  u16* vin  = (u16*)p; p += (i64)B_ * S_ * DKV_ * 2;
  u16* WqT  = (u16*)p; p += (i64)DQ_ * DQ_  * 2;
  u16* WkT  = (u16*)p; p += (i64)DQ_ * DKV_ * 2;
  u16* WvT  = (u16*)p; p += (i64)DQ_ * DKV_ * 2;
  u16* WoT  = (u16*)p; p += (i64)DQ_ * DQ_  * 2;
  u16* Qbf  = (u16*)p; p += (i64)B_ * T_ * DQ_ * 2;
  u16* Kbf  = (u16*)p; p += (i64)B_ * S_ * DQ_ * 2;
  u16* VtT  = (u16*)p; p += (i64)B_ * DQ_ * S_ * 2;  // [B,DQ,S]
  u16* Obf  = (u16*)p; p += (i64)B_ * T_ * DQ_ * 2;

  // 1) fp32 -> bf16 input copies
  {
    i64 n2q = (i64)B_ * T_ * DQ_ / 2;
    i64 n2k = (i64)B_ * S_ * DKV_ / 2;
    cvt_bf16<<<dim3((unsigned)((n2q + 255) / 256)), 256, 0, stream>>>(query, (unsigned int*)q_bf, n2q);
    cvt_bf16<<<dim3((unsigned)((n2k + 255) / 256)), 256, 0, stream>>>(key,   (unsigned int*)kin,  n2k);
    cvt_bf16<<<dim3((unsigned)((n2k + 255) / 256)), 256, 0, stream>>>(value, (unsigned int*)vin,  n2k);
  }
  // 2) weight transpose+convert: W[K,N] -> WT[N,K] bf16
  cvt_transpose<<<dim3((DQ_  * DQ_ + 255) / 256), 256, 0, stream>>>(Wq, WqT, DQ_,  DQ_);
  cvt_transpose<<<dim3((DKV_ * DQ_ + 255) / 256), 256, 0, stream>>>(Wk, WkT, DKV_, DQ_);
  cvt_transpose<<<dim3((DKV_ * DQ_ + 255) / 256), 256, 0, stream>>>(Wv, WvT, DKV_, DQ_);
  cvt_transpose<<<dim3((DQ_  * DQ_ + 255) / 256), 256, 0, stream>>>(Wo, WoT, DQ_,  DQ_);

  // 3) projections (M = B*T = B*S = 8192; grid 64 x 8 blocks of 128x64)
  gemm_bf16_nt<<<dim3(64, 8), 256, 0, stream>>>(q_bf, DQ_,  WqT, DQ_,  bq, DQ_,  Qbf, 1, DQ_, 0);
  gemm_bf16_nt<<<dim3(64, 8), 256, 0, stream>>>(kin,  DKV_, WkT, DKV_, bk, DKV_, Kbf, 1, DQ_, 0);
  gemm_bf16_nt<<<dim3(64, 8), 256, 0, stream>>>(vin,  DKV_, WvT, DKV_, bv, DKV_, VtT, 2, DQ_, S_);

  // 4) scores -> attn region of d_out (fp32, scaled + masked)
  attn_scores<<<dim3(T_ / 128, S_ / 64, B_ * H_), 256, 0, stream>>>(Qbf, Kbf, amask, kpm, attn);

  // 5) softmax in place over S (one block per (b,h,t) row)
  softmax_row<<<dim3((unsigned)((i64)B_ * H_ * T_)), 256, 0, stream>>>(attn);

  // 6) P @ V -> Obf (bf16 [B,T,DQ])
  attn_value<<<dim3(T_ / 128, B_ * H_), 256, 0, stream>>>(attn, VtT, Obf);

  // 7) final projection: out = Obf @ Wo + bo (fp32)
  gemm_bf16_nt<<<dim3(64, 8), 256, 0, stream>>>(Obf, DQ_, WoT, DQ_, bo, DQ_, out, 0, DQ_, 0);
}
